// MemoryGraph_7464653160919
// MI455X (gfx1250) — matrix operand order repaired and checked by reading, not compiled, not run on hardware
//
#include <hip/hip_runtime.h>
#include <math.h>

// ---- problem constants (from reference setup_inputs) ----
#define BB 4
#define NN 2048
#define DD 64
#define TT 8
#define HS 256
#define HM 256
#define KC 64               // K chunk staged in LDS per iteration (double-buffered)
#define GAMMA_MAX 0.1f
#define EPSV 1e-6f

typedef __attribute__((ext_vector_type(16))) __bf16 v16bf;
typedef __attribute__((ext_vector_type(8)))  float  v8f;
typedef __attribute__((ext_vector_type(4)))  float  f32x4;

__device__ __forceinline__ float sigf(float x) { return 1.0f / (1.0f + __expf(-x)); }

// ---------------------------------------------------------------------------
// Kernel 1: message MLP  m = tanh(tanh(h @ w1^T + b1) @ w2^T + b2)
// one 64-thread block per (b,n) row; m emitted in bf16 for the WMMA GEMM
// ---------------------------------------------------------------------------
__global__ void msg_mlp_kernel(const float* __restrict__ h,    // [B,N,D]
                               const float* __restrict__ w1,   // [HM,D]
                               const float* __restrict__ b1,   // [HM]
                               const float* __restrict__ w2,   // [D,HM]
                               const float* __restrict__ b2,   // [D]
                               __bf16* __restrict__ m_out)     // [B,N,D] bf16
{
    const int row = blockIdx.x;           // 0 .. B*N-1
    const int tid = threadIdx.x;          // 0 .. 63
    __shared__ float sh[DD];
    __shared__ float shid[HM];

    sh[tid] = h[(size_t)row * DD + tid];
    __syncthreads();

    #pragma unroll
    for (int jj = 0; jj < 4; ++jj) {
        const int j = tid * 4 + jj;
        float acc = b1[j];
        const float* wr = w1 + (size_t)j * DD;
        #pragma unroll
        for (int k = 0; k < DD; ++k) acc += sh[k] * wr[k];
        shid[j] = tanhf(acc);
    }
    __syncthreads();

    float acc = b2[tid];
    const float* wr = w2 + (size_t)tid * HM;
    #pragma unroll 8
    for (int k = 0; k < HM; ++k) acc += shid[k] * wr[k];
    m_out[(size_t)row * DD + tid] = (__bf16)tanhf(acc);
}

// ---------------------------------------------------------------------------
// Kernel 2: agg[b] = W[b] (NxN, f32) @ m[b] (NxD, bf16)  via bf16 WMMA.
// block = 256 threads (8 waves) -> 64x64 output tile. Wave w owns 16 rows x
// 32 cols (rt = w>>1, cpair = w&1): one A fragment feeds TWO wmma ops.
// K chunked at KC=64 and DOUBLE-BUFFERED in LDS:
//   * W chunk (64xKC f32) arrives via global_load_async_to_lds_b128
//     (ASYNCcnt path); the next chunk's DMA is issued before computing the
//     current one, so global latency hides under the WMMA stream.
//   * m chunk staged TRANSPOSED (mT[col][k]) so each lane's B operand is one
//     contiguous 32B ds_load (B layout: lanes 0-15 K=0..15, 16-31 K=16..31).
// Rows padded (68 / 72) to spread 16-lane column reads across LDS banks.
// ---------------------------------------------------------------------------
__global__ void __launch_bounds__(256)
agg_wmma_kernel(const float*  __restrict__ W,  // [B,N,N]
                const __bf16* __restrict__ m,  // [B,N,D]
                float* __restrict__ agg)       // [B,N,D]
{
    constexpr int WPAD = KC + 4;   // f32 row stride: 272B, 16B aligned
    constexpr int MPAD = KC + 8;   // bf16 row stride: 144B, 16B aligned
    __shared__ float  Ws[2][64][WPAD];   // W chunks: 64 rows x KC, x2 buffers
    __shared__ __bf16 mT[2][DD][MPAD];   // m chunks transposed: 64 cols x KC

    const int b    = blockIdx.y;
    const int row0 = blockIdx.x * 64;
    const int tid  = threadIdx.x;
    const int wave = tid >> 5;            // 0..7
    const int lane = tid & 31;
    const int rt    = wave >> 1;          // 0..3  row tile (16 rows)
    const int cpair = wave & 1;           // 0..1  column pair (32 cols)
    const int ml   = lane & 15;
    const int hi   = lane >> 4;

    const float*  Wb = W + (size_t)b * NN * NN;
    const __bf16* mb = m + (size_t)b * NN * DD;

    // ---- staging helpers ----
    auto stageW = [&](int kb, int buf) {
        // 64 rows x KC f32 = 1024 x 16B async copies, fully coalesced
        #pragma unroll
        for (int j = 0; j < 4; ++j) {
            const int i  = tid + 256 * j;        // 0..1023
            const int r  = i >> 4;               // 0..63
            const int c4 = i & 15;               // 0..15 (float4 index)
            const float* gp = Wb + ((size_t)(row0 + r)) * NN + kb + c4 * 4;
            const unsigned lp = (unsigned)(uintptr_t)&Ws[buf][r][c4 * 4];
            asm volatile("global_load_async_to_lds_b128 %0, %1, off"
                         :: "v"(lp), "v"((unsigned long long)(uintptr_t)gp)
                         : "memory");
        }
    };
    auto stageM = [&](int kb, int buf) {
        // transpose KC x 64 bf16 into mT[buf][c][k]
        const int k    = tid >> 2;               // 0..63
        const int part = tid & 3;                // 16 cols each
        const __bf16* mp = mb + (size_t)(kb + k) * DD + part * 16;
        #pragma unroll
        for (int e = 0; e < 16; ++e) mT[buf][part * 16 + e][k] = mp[e];
    };

    // ---- prologue: stage chunk 0 ----
    stageW(0, 0);
    stageM(0, 0);
    asm volatile("s_wait_asynccnt 0x0" ::: "memory");
    __syncthreads();

    v8f acc0 = {}, acc1 = {};
    int cur = 0;
    for (int kb = 0; kb < NN; kb += KC) {
        const int nxt = cur ^ 1;
        // issue next chunk's DMA before computing the current one
        if (kb + KC < NN) {
            stageW(kb + KC, nxt);
            stageM(kb + KC, nxt);
        }

        // ---- compute current chunk: 2 K-steps x 2 WMMAs ----
        const int ar = rt * 16 + ml;             // A row in LDS
        #pragma unroll
        for (int kl = 0; kl < KC; kl += 32) {
            // A (16x32 bf16): group0 K = kl+{0..15}, group1 K = kl+{16..31}
            const f32x4* wp0 = (const f32x4*)&Ws[cur][ar][kl + hi * 16];
            const f32x4* wp1 = (const f32x4*)&Ws[cur][ar][kl + hi * 16 + 8];
            const f32x4 a0 = wp0[0], a1 = wp0[1], a2 = wp1[0], a3 = wp1[1];
            v16bf a;
            #pragma unroll
            for (int e = 0; e < 4; ++e) {
                a[e]      = (__bf16)a0[e];
                a[4 + e]  = (__bf16)a1[e];
                a[8 + e]  = (__bf16)a2[e];
                a[12 + e] = (__bf16)a3[e];
            }
            // B: one contiguous 32B LDS read per lane per column tile
            const v16bf b0 = *(const v16bf*)&mT[cur][cpair * 32 + ml][kl + hi * 16];
            const v16bf b1 = *(const v16bf*)&mT[cur][cpair * 32 + 16 + ml][kl + hi * 16];
            acc0 = __builtin_amdgcn_wmma_f32_16x16x32_bf16(false, a, false, b0,
                                                           (short)0, acc0, false, false);
            acc1 = __builtin_amdgcn_wmma_f32_16x16x32_bf16(false, a, false, b1,
                                                           (short)0, acc1, false, false);
        }

        // drain our async DMA for the next buffer, then block-wide sync
        asm volatile("s_wait_asynccnt 0x0" ::: "memory");
        __syncthreads();
        cur = nxt;
    }

    // C layout: VGPR v -> row (v + hi*8), col = ml
    float* out = agg + (size_t)b * NN * DD;
    const int orow = row0 + rt * 16 + hi * 8;
    const int ocol = cpair * 32 + ml;
    #pragma unroll
    for (int v = 0; v < 8; ++v) {
        out[(size_t)(orow + v) * DD + ocol]      = acc0[v];
        out[(size_t)(orow + v) * DD + ocol + 16] = acc1[v];
    }
}

// ---------------------------------------------------------------------------
// Kernel 3: state MLP + decay update + h update + activity a
// one 64-thread block per (b,n) row
// ---------------------------------------------------------------------------
__global__ void state_kernel(const float* __restrict__ x_t,   // [B,N,D]
                             float* __restrict__ h,           // [B,N,D] in/out
                             const float* __restrict__ agg,   // [B,N,D]
                             const float* __restrict__ w1,    // [HS,2D]
                             const float* __restrict__ b1,    // [HS]
                             const float* __restrict__ w2,    // [D,HS]
                             const float* __restrict__ b2,    // [D]
                             float* __restrict__ decay,       // [B,N] in/out
                             const float* __restrict__ dg_logit, // [N]
                             const float* __restrict__ nid,   // [N,D]
                             float* __restrict__ a_out)       // [B,N]
{
    const int row = blockIdx.x;           // b*N + n
    const int n   = row & (NN - 1);
    const int tid = threadIdx.x;          // 0..63
    __shared__ float sinp[2 * DD];
    __shared__ float shid[HS];
    __shared__ float sred[DD];

    const float hold  = h[(size_t)row * DD + tid];
    const float dprev = decay[row];       // read before any write (sync below)

    sinp[tid]      = hold + x_t[(size_t)row * DD + tid];
    sinp[DD + tid] = agg[(size_t)row * DD + tid];
    __syncthreads();

    #pragma unroll
    for (int jj = 0; jj < 4; ++jj) {
        const int j = tid * 4 + jj;
        float acc = b1[j];
        const float* wr = w1 + (size_t)j * (2 * DD);
        #pragma unroll
        for (int k = 0; k < 2 * DD; ++k) acc += sinp[k] * wr[k];
        shid[j] = tanhf(acc);
    }
    __syncthreads();

    float acc = b2[tid];
    const float* wr = w2 + (size_t)tid * HS;
    #pragma unroll 8
    for (int k = 0; k < HS; ++k) acc += shid[k] * wr[k];
    const float hc = tanhf(acc);

    sred[tid] = hc;
    __syncthreads();
    for (int s = 32; s > 0; s >>= 1) {
        if (tid < s) sred[tid] += sred[tid + s];
        __syncthreads();
    }
    const float mean = sred[0] * (1.0f / (float)DD);

    const float dgam = GAMMA_MAX * sigf(dg_logit[n]);
    const float dn   = (1.0f - dgam) * dprev + dgam * sigf(mean);
    const float hn   = dn * hold + (1.0f - dn) * hc;
    h[(size_t)row * DD + tid] = hn;

    __syncthreads();
    sred[tid] = hn * nid[(size_t)n * DD + tid];
    __syncthreads();
    for (int s = 32; s > 0; s >>= 1) {
        if (tid < s) sred[tid] += sred[tid + s];
        __syncthreads();
    }
    if (tid == 0) {
        a_out[row] = tanhf(sred[0] * 0.125f);   // 1/sqrt(64)
        decay[row] = dn;
    }
}

// ---------------------------------------------------------------------------
// Kernel 4 (bandwidth hog, L2-resident): Hebbian update + W decay + row rms
// one 256-thread block per (b,n) row; row read once, written once.
// ---------------------------------------------------------------------------
__global__ void hebw_kernel(const float* __restrict__ a,        // [B,N]
                            float* __restrict__ heb,            // [B,N,N] in/out
                            float* __restrict__ W,              // [B,N,N] in/out
                            const float* __restrict__ wg_logit, // [N]
                            const float* __restrict__ hg_logit) // [N]
{
    const int row = blockIdx.x;            // b*N + n
    const int b   = row >> 11;             // N = 2048
    const int n   = row & (NN - 1);
    const int tid = threadIdx.x;           // 0..255

    const float an = a[row];
    const float hg = GAMMA_MAX * sigf(hg_logit[n]);
    const float wg = GAMMA_MAX * sigf(wg_logit[n]);
    const size_t base = (size_t)row * NN;
    const float* arow = a + (size_t)b * NN;

    float wtmp[NN / 256];
    float ss = 0.0f;
    #pragma unroll
    for (int i = 0; i < NN / 256; ++i) {
        const int j  = tid + i * 256;
        const float hb = (1.0f - hg) * heb[base + j] + hg * (an * arow[j]);
        heb[base + j]  = hb;
        const float wt = (1.0f - wg) * W[base + j] + wg * hb;
        wtmp[i] = wt;
        ss += wt * wt;
    }

    __shared__ float sred[256];
    sred[tid] = ss;
    __syncthreads();
    for (int s = 128; s > 0; s >>= 1) {
        if (tid < s) sred[tid] += sred[tid + s];
        __syncthreads();
    }
    const float scale = rsqrtf(sred[0] * (1.0f / (float)NN) + EPSV);

    #pragma unroll
    for (int i = 0; i < NN / 256; ++i)
        W[base + tid + i * 256] = wtmp[i] * scale;
}

// ---------------------------------------------------------------------------
// Host orchestration
// ---------------------------------------------------------------------------
extern "C" void kernel_launch(void* const* d_in, const int* in_sizes, int n_in,
                              void* d_out, int out_size, void* d_ws, size_t ws_size,
                              hipStream_t stream) {
    const float* x        = (const float*)d_in[0];   // [T,B,N,D]
    const float* h0       = (const float*)d_in[1];   // [B,N,D]
    // d_in[2] = msg carry: never read by the recurrence, output is h only
    const float* W0       = (const float*)d_in[3];   // [B,N,N]
    const float* heb0     = (const float*)d_in[4];   // [B,N,N]
    const float* decay0   = (const float*)d_in[5];   // [B,N]
    const float* nid      = (const float*)d_in[6];   // [N,D]
    const float* sw1      = (const float*)d_in[7];   // [HS,2D]
    const float* sb1      = (const float*)d_in[8];
    const float* sw2      = (const float*)d_in[9];   // [D,HS]
    const float* sb2      = (const float*)d_in[10];
    const float* mw1      = (const float*)d_in[11];  // [HM,D]
    const float* mb1      = (const float*)d_in[12];
    const float* mw2      = (const float*)d_in[13];  // [D,HM]
    const float* mb2      = (const float*)d_in[14];
    const float* wg_logit = (const float*)d_in[15];  // [N]
    const float* dg_logit = (const float*)d_in[16];  // [N]
    const float* hg_logit = (const float*)d_in[17];  // [N]

    const size_t nnB  = (size_t)BB * NN * NN * sizeof(float);  // 67.1 MB
    const size_t ndB  = (size_t)BB * NN * DD * sizeof(float);  //  2.1 MB
    const size_t nB   = (size_t)BB * NN * sizeof(float);

    char* p = (char*)d_ws;
    float*  W     = (float*)p;  p += nnB;
    float*  heb   = (float*)p;  p += nnB;
    float*  h     = (float*)p;  p += ndB;
    float*  agg   = (float*)p;  p += ndB;
    float*  decay = (float*)p;  p += nB;
    float*  aact  = (float*)p;  p += nB;
    __bf16* mbf   = (__bf16*)p; p += (size_t)BB * NN * DD * sizeof(__bf16);

    hipMemcpyAsync(W,     W0,     nnB, hipMemcpyDeviceToDevice, stream);
    hipMemcpyAsync(heb,   heb0,   nnB, hipMemcpyDeviceToDevice, stream);
    hipMemcpyAsync(h,     h0,     ndB, hipMemcpyDeviceToDevice, stream);
    hipMemcpyAsync(decay, decay0, nB,  hipMemcpyDeviceToDevice, stream);

    const int rows = BB * NN;                 // 8192
    const dim3 aggGrid(NN / 64, BB);          // (32, 4), 8 waves/block

    for (int t = 0; t < TT; ++t) {
        const float* x_t = x + (size_t)t * BB * NN * DD;

        msg_mlp_kernel<<<rows, 64, 0, stream>>>(h, mw1, mb1, mw2, mb2, mbf);
        agg_wmma_kernel<<<aggGrid, 256, 0, stream>>>(W, mbf, agg);
        state_kernel<<<rows, 64, 0, stream>>>(x_t, h, agg, sw1, sb1, sw2, sb2,
                                              decay, dg_logit, nid, aact);
        hebw_kernel<<<rows, 256, 0, stream>>>(aact, heb, W, wg_logit, hg_logit);
    }

    hipMemcpyAsync(d_out, h, ndB, hipMemcpyDeviceToDevice, stream);
}
